// TLSTMCell_60455959659035
// MI455X (gfx1250) — matrix operand order repaired and checked by reading, not compiled, not run on hardware
//
#include <hip/hip_runtime.h>

typedef __bf16 bf_t;
typedef __attribute__((ext_vector_type(16))) __bf16 v16bf;
typedef __attribute__((ext_vector_type(8)))  __bf16 v8bf;
typedef __attribute__((ext_vector_type(8)))  float  v8f;

#define BATCH 8192
#define ISZ   1024
#define HSZ   1024
#define KCOMB 2048   // ISZ + HSZ
#define NGATE 4096   // 4*HSZ
#define LDA_S 40     // LDS row stride in bf16 (80 B, 16B-aligned, pad vs banks)

// round-to-nearest-even f32 -> bf16
__device__ __forceinline__ bf_t f2bf(float f) {
  union { float f; unsigned u; } in; in.f = f;
  unsigned u = in.u;
  unsigned r = (u + 0x7fffu + ((u >> 16) & 1u)) >> 16;
  union { unsigned short s; bf_t b; } out; out.s = (unsigned short)r;
  return out.b;
}

__device__ __forceinline__ float sigmoidf_(float x) {
  return 1.0f / (1.0f + __expf(-x));
}

// --- CDNA5 async global->LDS copy (ASYNCcnt path) ---------------------------
__device__ __forceinline__ void async_ld_b128(unsigned lds_off, const bf_t* g) {
  asm volatile("global_load_async_to_lds_b128 %0, %1, off"
               :: "v"(lds_off), "v"((unsigned long long)(uintptr_t)g)
               : "memory");
}
__device__ __forceinline__ void wait_async0() {
  asm volatile("s_wait_asynccnt 0x0" ::: "memory");
}

// issue one 128x32 bf16 tile: 256 threads x 2 async b128 = 8192 B
__device__ __forceinline__ void issue_tile(const bf_t* __restrict__ Abase, size_t lda,
                                           int Mblk, int kk, bf_t* buf, int tid) {
  int g0 = tid, g1 = tid + 256;
  int r0 = g0 >> 2, c0 = (g0 & 3) * 8;
  int r1 = g1 >> 2, c1 = (g1 & 3) * 8;
  unsigned l0 = (unsigned)(uintptr_t)&buf[r0 * LDA_S + c0];
  unsigned l1 = (unsigned)(uintptr_t)&buf[r1 * LDA_S + c1];
  async_ld_b128(l0, Abase + (size_t)(Mblk + r0) * lda + kk + c0);
  async_ld_b128(l1, Abase + (size_t)(Mblk + r1) * lda + kk + c1);
}

// build a 16x32 A-fragment from an LDS row block (row = frag row base + lane&15)
__device__ __forceinline__ v16bf a_frag(const bf_t* buf, int rowbase, int lane) {
  int ml = rowbase + (lane & 15);
  int kb = (lane >> 4) * 16;
  v8bf lo = *(const v8bf*)(&buf[ml * LDA_S + kb]);
  v8bf hi = *(const v8bf*)(&buf[ml * LDA_S + kb + 8]);
  v16bf a;
#pragma unroll
  for (int e = 0; e < 8; ++e) { a[e] = lo[e]; a[e + 8] = hi[e]; }
  return a;
}

// ---------------------------------------------------------------------------
// Kernel 1: convert weights to bf16 (row-major [n][k] == WMMA B-fragment order)
// ---------------------------------------------------------------------------
__global__ void convert_weights(const float* __restrict__ Wl, const float* __restrict__ Wd,
                                bf_t* __restrict__ Wl_bf, bf_t* __restrict__ Wd_bf) {
  size_t i = (size_t)blockIdx.x * blockDim.x + threadIdx.x;
  const size_t nl = (size_t)NGATE * KCOMB;
  const size_t nd = (size_t)HSZ * ISZ;
  if (i < nl) Wl_bf[i] = f2bf(Wl[i]);
  if (i < nd) Wd_bf[i] = f2bf(Wd[i]);
}

// ---------------------------------------------------------------------------
// Kernel 2: x -> combined_bf[:, :1024], h_cur -> h_bf, T[m] = 1/log(dt + e)
// ---------------------------------------------------------------------------
__global__ void prep_inputs(const float* __restrict__ input, const float* __restrict__ h_cur,
                            bf_t* __restrict__ comb_bf, bf_t* __restrict__ h_bf,
                            float* __restrict__ T) {
  size_t i = (size_t)blockIdx.x * blockDim.x + threadIdx.x;
  int m = (int)(i >> 10);
  int j = (int)(i & 1023);
  const float* row = input + (size_t)m * (ISZ + 1);
  comb_bf[(size_t)m * KCOMB + j] = f2bf(row[j]);
  h_bf[i] = f2bf(h_cur[i]);
  if (j == 0) T[m] = 1.0f / __logf(row[ISZ] + 2.7183f);
}

// ---------------------------------------------------------------------------
// Kernel 3: GEMM1 (h_cur @ W_desc^T) + h_adj epilogue -> combined_bf[:,1024:]
// Block: 8 waves = 4(M) x 2(N); wave tile 32(M) x 16(N); block 128 x 32.
// Double-buffered async LDS A tiles, one barrier per K-step.
// ---------------------------------------------------------------------------
__global__ __launch_bounds__(256)
void gemm1_hadj(const bf_t* __restrict__ A,    // h_bf [8192][1024]
                const bf_t* __restrict__ Wd,   // [1024][1024] (n,k)
                const float* __restrict__ bd,
                const float* __restrict__ h_cur,
                const float* __restrict__ T,
                bf_t* __restrict__ comb_bf) {  // [8192][2048]
  __shared__ bf_t As[2][128 * LDA_S];
  const int tid   = threadIdx.x;
  const int lane  = tid & 31;
  const int wid   = tid >> 5;
  const int waveM = wid >> 1;                  // 0..3 -> 32 rows each
  const int waveN = wid & 1;                   // 0..1
  const int Mblk  = blockIdx.y * 128;
  const int n     = blockIdx.x * 32 + waveN * 16 + (lane & 15);
  const int k0b   = (lane >> 4) * 16;
  const int row0  = waveM * 32;

  const int iters = ISZ / 32;                  // 32
  issue_tile(A, ISZ, Mblk, 0, As[0], tid);

  v8f acc0 = {}, acc1 = {};
  for (int t = 0; t < iters; ++t) {
    const int kk = t * 32;
    wait_async0();
    __syncthreads();
    if (t + 1 < iters) issue_tile(A, ISZ, Mblk, kk + 32, As[(t + 1) & 1], tid);

    const bf_t* buf = As[t & 1];
    v16bf a0 = a_frag(buf, row0, lane);
    v16bf a1 = a_frag(buf, row0 + 16, lane);
    v16bf b  = *(const v16bf*)(Wd + (size_t)n * ISZ + kk + k0b);

    acc0 = __builtin_amdgcn_wmma_f32_16x16x32_bf16(false, a0, false, b, (short)0, acc0, false, false);
    acc1 = __builtin_amdgcn_wmma_f32_16x16x32_bf16(false, a1, false, b, (short)0, acc1, false, false);
  }

  const float bias  = bd[n];
  const int   mrow0 = Mblk + row0 + 8 * (lane >> 4);
#pragma unroll
  for (int r = 0; r < 8; ++r) {
    int m0 = mrow0 + r;
    int m1 = m0 + 16;
    float h0 = h_cur[(size_t)m0 * HSZ + n] + (T[m0] - 1.0f) * tanhf(acc0[r] + bias);
    float h1 = h_cur[(size_t)m1 * HSZ + n] + (T[m1] - 1.0f) * tanhf(acc1[r] + bias);
    comb_bf[(size_t)m0 * KCOMB + ISZ + n] = f2bf(h0);
    comb_bf[(size_t)m1 * KCOMB + ISZ + n] = f2bf(h1);
  }
}

// ---------------------------------------------------------------------------
// Kernel 4: GEMM2 (combined @ W_layers^T), 4 gates + 2 M-frags per wave,
// fused LSTM epilogue -> h_next, c_next.
// ---------------------------------------------------------------------------
__global__ __launch_bounds__(256)
void gemm2_lstm(const bf_t* __restrict__ A,    // comb_bf [8192][2048]
                const bf_t* __restrict__ Wl,   // [4096][2048]
                const float* __restrict__ bl,
                const float* __restrict__ c_cur,
                float* __restrict__ out) {
  __shared__ bf_t As[2][128 * LDA_S];
  const int tid   = threadIdx.x;
  const int lane  = tid & 31;
  const int wid   = tid >> 5;
  const int waveM = wid >> 1;
  const int waveN = wid & 1;
  const int Mblk  = blockIdx.y * 128;
  const int n     = blockIdx.x * 32 + waveN * 16 + (lane & 15);
  const int k0b   = (lane >> 4) * 16;
  const int row0  = waveM * 32;

  const bf_t* Bp[4] = { Wl + (size_t)(n)           * KCOMB,
                        Wl + (size_t)(n +     HSZ) * KCOMB,
                        Wl + (size_t)(n + 2 * HSZ) * KCOMB,
                        Wl + (size_t)(n + 3 * HSZ) * KCOMB };

  const int iters = KCOMB / 32;                // 64
  issue_tile(A, KCOMB, Mblk, 0, As[0], tid);

  v8f acc[4][2] = {};
  for (int t = 0; t < iters; ++t) {
    const int kk = t * 32;
    wait_async0();
    __syncthreads();
    if (t + 1 < iters) issue_tile(A, KCOMB, Mblk, kk + 32, As[(t + 1) & 1], tid);

    const bf_t* buf = As[t & 1];
    v16bf a0 = a_frag(buf, row0, lane);
    v16bf a1 = a_frag(buf, row0 + 16, lane);
#pragma unroll
    for (int g = 0; g < 4; ++g) {
      v16bf b = *(const v16bf*)(Bp[g] + kk + k0b);
      acc[g][0] = __builtin_amdgcn_wmma_f32_16x16x32_bf16(false, a0, false, b, (short)0, acc[g][0], false, false);
      acc[g][1] = __builtin_amdgcn_wmma_f32_16x16x32_bf16(false, a1, false, b, (short)0, acc[g][1], false, false);
    }
  }

  const float bi_ = bl[n];
  const float bf_ = bl[n + HSZ];
  const float bo_ = bl[n + 2 * HSZ];
  const float bg_ = bl[n + 3 * HSZ];
  const int   mrow0 = Mblk + row0 + 8 * (lane >> 4);
  const size_t coff = (size_t)BATCH * HSZ;
#pragma unroll
  for (int half = 0; half < 2; ++half) {
#pragma unroll
    for (int r = 0; r < 8; ++r) {
      int    m  = mrow0 + half * 16 + r;
      size_t ix = (size_t)m * HSZ + n;
      float gi = sigmoidf_(acc[0][half][r] + bi_);
      float gf = sigmoidf_(acc[1][half][r] + bf_);
      float go = sigmoidf_(acc[2][half][r] + bo_);
      float gg = tanhf(acc[3][half][r] + bg_);
      float c  = gf * c_cur[ix] + gi * gg;
      float h  = go * tanhf(c);
      out[ix]        = h;
      out[coff + ix] = c;
    }
  }
}

// ---------------------------------------------------------------------------
extern "C" void kernel_launch(void* const* d_in, const int* in_sizes, int n_in,
                              void* d_out, int out_size, void* d_ws, size_t ws_size,
                              hipStream_t stream) {
  const float* input = (const float*)d_in[0];
  const float* h_cur = (const float*)d_in[1];
  const float* c_cur = (const float*)d_in[2];
  const float* Wl    = (const float*)d_in[3];
  const float* bl    = (const float*)d_in[4];
  const float* Wd    = (const float*)d_in[5];
  const float* bd    = (const float*)d_in[6];
  float* out = (float*)d_out;

  char* ws = (char*)d_ws;
  bf_t*  Wl_bf   = (bf_t*)(ws);                    // 16,777,216 B
  bf_t*  Wd_bf   = (bf_t*)(ws + 16777216);         //  2,097,152 B
  bf_t*  h_bf    = (bf_t*)(ws + 18874368);         // 16,777,216 B
  bf_t*  comb_bf = (bf_t*)(ws + 35651584);         // 33,554,432 B
  float* T       = (float*)(ws + 69206016);        //     32,768 B

  convert_weights<<<32768, 256, 0, stream>>>(Wl, Wd, Wl_bf, Wd_bf);
  prep_inputs<<<32768, 256, 0, stream>>>(input, h_cur, comb_bf, h_bf, T);

  dim3 g1(HSZ / 32, BATCH / 128);   // (32, 64)
  gemm1_hadj<<<g1, 256, 0, stream>>>(h_bf, Wd_bf, bd, h_cur, T, comb_bf);

  dim3 g2(HSZ / 32, BATCH / 128);   // (32, 64)
  gemm2_lstm<<<g2, 256, 0, stream>>>(comb_bf, Wl_bf, bl, c_cur, out);
}
